// model_34995393528275
// MI455X (gfx1250) — compile-verified, matrix operand order checked by reading, not run on hardware
//
#include <hip/hip_runtime.h>
#include <hip/hip_bf16.h>

typedef __attribute__((ext_vector_type(16))) _Float16 v16h;
typedef __attribute__((ext_vector_type(8)))  float    v8f;

#define HWSZ 2304   // 48*48
#define NTOT 4608   // B*HW
#define EPSV 1e-5f

__device__ __forceinline__ float siluf(float v){ return v / (1.0f + expf(-v)); }

// ---------------------------------------------------------------------------
// Generic WMMA GEMM: C[m][n] (+bias, act) = sum_k A[m][k]*Bt[n][k]
// A : f16 [Mreal x K] row-major (zero-extended rows beyond Mreal)
// Bt: f16 [N x K] row-major (pixel-major activations -> contiguous b128 loads)
// C : f32 [Mpad x N] (may be null). ChT: f16 [N x ldch] transposed mirror
// (feeds the next GEMM's B operand directly); null to skip.
// One wave per 16 x (16*NT) output tile. Batched via blockIdx.z.
// act: 0 none, 1 silu.
// ---------------------------------------------------------------------------
template<int NT>
__global__ __launch_bounds__(32)
void wmma_gemm(const _Float16* __restrict__ A, const _Float16* __restrict__ Bt,
               float* __restrict__ C, _Float16* __restrict__ ChT, int ldch,
               const float* __restrict__ bias,
               int Mreal, int K, int N, int act,
               long sA, long sB, long sC, int Amod)
{
    int lane = threadIdx.x;
    int nt = blockIdx.x, mt = blockIdx.y, z = blockIdx.z;
    const _Float16* Ab = A + (long)(z % Amod) * sA;
    const _Float16* Bb = Bt + (long)z * sB;
    long cOff = (long)z * sC;
    int lm = lane & 15, hi = lane >> 4;
    int m = mt * 16 + lm;
    int nBase = nt * 16 * NT;
    v8f acc[NT];
    #pragma unroll
    for (int j = 0; j < NT; ++j) acc[j] = v8f{};
    for (int k0 = 0; k0 < K; k0 += 32) {
        v16h af;
        if (m < Mreal) {
            const _Float16* r = Ab + (long)m * K + k0 + hi * 8;
            #pragma unroll
            for (int i = 0; i < 8; ++i) { af[i] = r[i]; af[i + 8] = r[16 + i]; }
        } else {
            #pragma unroll
            for (int i = 0; i < 16; ++i) af[i] = (_Float16)0.0f;
        }
        #pragma unroll
        for (int j = 0; j < NT; ++j) {
            int n = nBase + j * 16 + lm;
            const _Float16* p = Bb + (long)n * K + k0 + hi * 8;
            v16h bf;
            #pragma unroll
            for (int i = 0; i < 8; ++i) { bf[i] = p[i]; bf[i + 8] = p[16 + i]; }
            acc[j] = __builtin_amdgcn_wmma_f32_16x16x32_f16(false, af, false, bf, (short)0, acc[j], false, false);
        }
    }
    #pragma unroll
    for (int j = 0; j < NT; ++j) {
        int n = nBase + j * 16 + lm;
        #pragma unroll
        for (int r = 0; r < 8; ++r) {
            int mm = mt * 16 + hi * 8 + r;
            float v = acc[j][r];
            if (bias) v += bias[mm < Mreal ? mm : 0];
            if (act == 1) v = siluf(v);
            if (C)   C[cOff + (long)mm * N + n] = v;
            if (ChT) ChT[cOff + (long)n * ldch + mm] = (_Float16)v;
        }
    }
}

// ---------------------------------------------------------------------------
// Implicit-GEMM 3x3 conv 640->64, pad 1, on 48x48.
// Wt : f16 [9][64][640] (tap-major weights)
// act: f16 [NTOT][640]  (pixel-major BN+ReLU'd concat)
// out: f32 [64][NTOT]
// One wave per 16 x 64 output tile.
// ---------------------------------------------------------------------------
__global__ __launch_bounds__(32)
void wmma_conv3x3(const _Float16* __restrict__ Wt, const _Float16* __restrict__ act,
                  float* __restrict__ out)
{
    int lane = threadIdx.x;
    int nt = blockIdx.x;   // 0..35  (N sub-blocks of 64)
    int mt = blockIdx.y;   // 0..3
    int b  = blockIdx.z;   // 0..1
    int lm = lane & 15, hi = lane >> 4;
    int m = mt * 16 + lm;
    v8f acc[4];
    #pragma unroll
    for (int j = 0; j < 4; ++j) acc[j] = v8f{};
    for (int t = 0; t < 9; ++t) {
        int dy = t / 3 - 1, dx = t % 3 - 1;
        bool okj[4]; long colj[4];
        #pragma unroll
        for (int j = 0; j < 4; ++j) {
            int p = nt * 64 + j * 16 + lm;
            int h = p / 48, w = p % 48;
            int hh = h + dy, ww = w + dx;
            okj[j] = (hh >= 0 && hh < 48 && ww >= 0 && ww < 48);
            colj[j] = (long)(b * HWSZ + hh * 48 + ww) * 640;
        }
        const _Float16* Wrow = Wt + (long)t * 64 * 640 + (long)m * 640;
        for (int k0 = 0; k0 < 640; k0 += 32) {
            v16h af;
            {
                const _Float16* r = Wrow + k0 + hi * 8;
                #pragma unroll
                for (int i = 0; i < 8; ++i) { af[i] = r[i]; af[i + 8] = r[16 + i]; }
            }
            #pragma unroll
            for (int j = 0; j < 4; ++j) {
                v16h bf;
                if (okj[j]) {
                    const _Float16* p = act + colj[j] + k0 + hi * 8;
                    #pragma unroll
                    for (int i = 0; i < 8; ++i) { bf[i] = p[i]; bf[i + 8] = p[16 + i]; }
                } else {
                    #pragma unroll
                    for (int i = 0; i < 16; ++i) bf[i] = (_Float16)0.0f;
                }
                acc[j] = __builtin_amdgcn_wmma_f32_16x16x32_f16(false, af, false, bf, (short)0, acc[j], false, false);
            }
        }
    }
    #pragma unroll
    for (int j = 0; j < 4; ++j) {
        int n = nt * 64 + j * 16 + lm;
        #pragma unroll
        for (int r = 0; r < 8; ++r) {
            int mm = mt * 16 + hi * 8 + r;
            out[(long)mm * NTOT + (long)b * HWSZ + n] = acc[j][r];
        }
    }
}

// ------------------------- elementwise / prep kernels -----------------------
__global__ void cvt16(const float* __restrict__ in, _Float16* __restrict__ out, int n)
{
    int i = blockIdx.x * blockDim.x + threadIdx.x;
    if (i < n) out[i] = (_Float16)in[i];
}

// BN(eval)+ReLU on x (B,C,H,W) -> f16 [NTOT][64] pixel-major
__global__ void bnrelu_x(const float* __restrict__ x, const float* __restrict__ g,
                         const float* __restrict__ bb, _Float16* __restrict__ out)
{
    int i = blockIdx.x * blockDim.x + threadIdx.x;
    if (i >= 64 * NTOT) return;
    int c = i / NTOT, bp = i % NTOT, b = bp / HWSZ, p = bp % HWSZ;
    float rs = rsqrtf(1.0f + EPSV);
    float v = x[((long)b * 64 + c) * HWSZ + p] * (g[c] * rs) + bb[c];
    out[(long)bp * 64 + c] = (_Float16)fmaxf(v, 0.0f);
}

// AvgPool (count_include_pad) + BN + ReLU -> f16 [B*So*So][64] pixel-major
__global__ void pool_bnrelu(const float* __restrict__ x, const float* __restrict__ g,
                            const float* __restrict__ bb, _Float16* __restrict__ out,
                            int ks, int stride, int pad, int So, int total)
{
    int i = blockIdx.x * blockDim.x + threadIdx.x;
    if (i >= total) return;
    int per = 2 * So * So;
    int c = i / per, rem = i % per, b = rem / (So * So), q = rem % (So * So);
    int oh = q / So, ow = q % So;
    float s = 0.0f;
    for (int ky = 0; ky < ks; ++ky)
        for (int kx = 0; kx < ks; ++kx) {
            int hh = oh * stride - pad + ky, ww = ow * stride - pad + kx;
            if (hh >= 0 && hh < 48 && ww >= 0 && ww < 48)
                s += x[((long)b * 64 + c) * HWSZ + hh * 48 + ww];
        }
    s /= (float)(ks * ks);
    float rs = rsqrtf(1.0f + EPSV);
    s = s * (g[c] * rs) + bb[c];
    out[(long)rem * 64 + c] = (_Float16)fmaxf(s, 0.0f);
}

// global-mean branch: z4[b][o] = sum_c relu(bn(mean_c)) * s4_w[o][c]
__global__ __launch_bounds__(128)
void mean_s4(const float* __restrict__ x, const float* __restrict__ g,
             const float* __restrict__ bb, const float* __restrict__ w,
             float* __restrict__ z4)
{
    __shared__ float sm[64];
    int b = blockIdx.x, t = threadIdx.x;
    if (t < 64) {
        const float* px = x + ((long)b * 64 + t) * HWSZ;
        float s = 0.0f;
        for (int p = 0; p < HWSZ; ++p) s += px[p];
        s /= (float)HWSZ;
        float rs = rsqrtf(1.0f + EPSV);
        sm[t] = fmaxf(s * (g[t] * rs) + bb[t], 0.0f);
    }
    __syncthreads();
    float acc = 0.0f;
    for (int c = 0; c < 64; ++c) acc += sm[c] * w[t * 64 + c];
    z4[b * 128 + t] = acc;
}

__device__ float bilin(const float* __restrict__ img, int S, int h, int w)
{
    float scale = (float)S / 48.0f;
    float fy = (h + 0.5f) * scale - 0.5f;
    float fx = (w + 0.5f) * scale - 0.5f;
    int y0 = (int)floorf(fy); float dy = fy - (float)y0;
    int x0 = (int)floorf(fx); float dx = fx - (float)x0;
    int y0c = min(max(y0, 0), S - 1), y1c = min(max(y0 + 1, 0), S - 1);
    int x0c = min(max(x0, 0), S - 1), x1c = min(max(x0 + 1, 0), S - 1);
    float v00 = img[y0c * S + x0c], v01 = img[y0c * S + x1c];
    float v10 = img[y1c * S + x0c], v11 = img[y1c * S + x1c];
    return v00 * (1 - dy) * (1 - dx) + v01 * (1 - dy) * dx + v10 * dy * (1 - dx) + v11 * dy * dx;
}

// Assemble 5-branch concat + comp BN+ReLU -> f16 [NTOT][640] pixel-major
__global__ void assemble_cat(const float* __restrict__ z0, const float* __restrict__ z1,
                             const float* __restrict__ z2, const float* __restrict__ z3,
                             const float* __restrict__ z4, const float* __restrict__ cg,
                             const float* __restrict__ cb, _Float16* __restrict__ cat)
{
    int i = blockIdx.x * blockDim.x + threadIdx.x;
    if (i >= 128 * NTOT) return;
    int o = i / NTOT, bp = i % NTOT, b = bp / HWSZ, p = bp % HWSZ;
    int h = p / 48, w = p % 48;
    float x0v = z0[i];
    float vals[5];
    vals[0] = x0v;
    vals[1] = z1[i] + x0v;
    vals[2] = bilin(z2 + (long)o * 1152 + b * 576, 24, h, w) + x0v;
    vals[3] = bilin(z3 + (long)o * 288 + b * 144, 12, h, w) + x0v;
    vals[4] = z4[b * 128 + o] + x0v;
    float rs = rsqrtf(1.0f + EPSV);
    #pragma unroll
    for (int j = 0; j < 5; ++j) {
        int cc = j * 128 + o;
        float v = vals[j] * (cg[cc] * rs) + cb[cc];
        cat[(long)bp * 640 + cc] = (_Float16)fmaxf(v, 0.0f);
    }
}

// reorder comp_w (O,IC,3,3) f32 -> [tap][O][IC] f16
__global__ void prep_compw(const float* __restrict__ w, _Float16* __restrict__ out)
{
    int i = blockIdx.x * blockDim.x + threadIdx.x;
    if (i >= 9 * 64 * 640) return;
    int t = i / (64 * 640), rem = i % (64 * 640), o = rem / 640, ic = rem % 640;
    int ky = t / 3, kx = t % 3;
    out[i] = (_Float16)w[((long)(o * 640 + ic) * 3 + ky) * 3 + kx];
}

// p = comp_out + shortcut -> f16 [NTOT][64] pixel-major
__global__ void make_ph(const float* __restrict__ comp, const float* __restrict__ zsc,
                        _Float16* __restrict__ ph)
{
    int i = blockIdx.x * blockDim.x + threadIdx.x;
    if (i >= 64 * NTOT) return;
    int c = i / NTOT, bp = i % NTOT;
    ph[(long)bp * 64 + c] = (_Float16)(comp[i] + zsc[i]);
}

// depthwise 3x3 + bias + SiLU on xz rows 0..127 -> xcs f32 [128][NTOT]
__global__ void dwconv_silu(const float* __restrict__ xz, const float* __restrict__ dww,
                            const float* __restrict__ dwb, float* __restrict__ xcs)
{
    int i = blockIdx.x * blockDim.x + threadIdx.x;
    if (i >= 128 * NTOT) return;
    int d = i / NTOT, bp = i % NTOT, b = bp / HWSZ, p = bp % HWSZ;
    int h = p / 48, w = p % 48;
    float v = dwb[d];
    #pragma unroll
    for (int ky = 0; ky < 3; ++ky)
        for (int kx = 0; kx < 3; ++kx) {
            int hh = h + ky - 1, ww = w + kx - 1;
            if (hh >= 0 && hh < 48 && ww >= 0 && ww < 48)
                v += xz[(long)d * NTOT + (long)b * HWSZ + hh * 48 + ww] * dww[d * 9 + ky * 3 + kx];
        }
    xcs[i] = siluf(v);
}

__device__ __forceinline__ int dir_pixel(int k, int l)
{
    if (k == 0) return l;
    if (k == 1) return (l % 48) * 48 + l / 48;
    if (k == 2) return 2303 - l;
    int r = 2303 - l; return (r % 48) * 48 + r / 48;
}

// build xs f16 [B*K][L][128] pixel-major (contiguous writes)
__global__ void xs_fill(const float* __restrict__ xcs, _Float16* __restrict__ xsh)
{
    int i = blockIdx.x * blockDim.x + threadIdx.x;
    if (i >= 8 * HWSZ * 128) return;
    int d = i % 128, l = (i / 128) % HWSZ, bk = i / (128 * HWSZ);
    int b = bk >> 2, k = bk & 3;
    xsh[i] = (_Float16)xcs[(long)d * NTOT + (long)b * HWSZ + dir_pixel(k, l)];
}

// delta = softplus(dt_w @ dts + dt_b) : f32 [B][K][128][L]
__global__ void delta_k(const float* __restrict__ xdbl, const float* __restrict__ dtw,
                        const float* __restrict__ dtb, float* __restrict__ delta)
{
    int i = blockIdx.x * blockDim.x + threadIdx.x;
    if (i >= 2 * 4 * 128 * HWSZ) return;
    int l = i % HWSZ, d = (i / HWSZ) % 128, k = (i / (HWSZ * 128)) % 4, b = i / (HWSZ * 128 * 4);
    const float* row = xdbl + (long)(b * 4 + k) * 48 * HWSZ;
    float s = dtb[k * 128 + d];
    #pragma unroll
    for (int r = 0; r < 4; ++r) s += dtw[(k * 128 + d) * 4 + r] * row[(long)r * HWSZ + l];
    delta[i] = (s > 20.0f) ? s : log1pf(expf(s));
}

__global__ void acoef_k(const float* __restrict__ alogs, float* __restrict__ ac)
{
    int i = blockIdx.x * blockDim.x + threadIdx.x;
    if (i < 4 * 128 * 16) ac[i] = -expf(alogs[i]);
}

// sequential selective scan: 16 state-lanes per chain, shuffle-reduced
__global__ __launch_bounds__(512)
void scan_k(const float* __restrict__ delta, const float* __restrict__ xdbl,
            const float* __restrict__ xcs, const float* __restrict__ ac,
            const float* __restrict__ Ds, float* __restrict__ y)
{
    int t = threadIdx.x;
    int n = t & 15, dl = t >> 4;          // 32 chains per block
    int bk = blockIdx.x;                  // b*4+k
    int b = bk >> 2, k = bk & 3;
    int d = blockIdx.y * 32 + dl;
    const float* del  = delta + ((long)bk * 128 + d) * HWSZ;
    const float* Bp   = xdbl + ((long)bk * 48 + 4 + n) * HWSZ;
    const float* Cp   = xdbl + ((long)bk * 48 + 20 + n) * HWSZ;
    const float* xrow = xcs + (long)d * NTOT + (long)b * HWSZ;
    float Ac = ac[(k * 128 + d) * 16 + n];
    float Dv = Ds[k * 128 + d];
    float* yrow = y + ((long)bk * 128 + d) * HWSZ;
    float hst = 0.0f;
    for (int l = 0; l < HWSZ; ++l) {
        float dv = del[l];
        float xv = xrow[dir_pixel(k, l)];
        float a = expf(dv * Ac);
        hst = a * hst + dv * xv * Bp[l];
        float c = hst * Cp[l];
        for (int o = 8; o > 0; o >>= 1) c += __shfl_xor(c, o, 16);
        if (n == 0) yrow[l] = c + xv * Dv;
    }
}

// combine 4 directions, LayerNorm over d, gate with silu(z) -> f16 [NTOT][128]
__global__ __launch_bounds__(128)
void combine_ln(const float* __restrict__ y, const float* __restrict__ xz,
                const float* __restrict__ lng, const float* __restrict__ lnb,
                _Float16* __restrict__ gh)
{
    __shared__ float sm[4], sq[4];
    int bl = blockIdx.x;
    int b = bl / HWSZ, l = bl % HWSZ;
    int d = threadIdx.x;
    int h = l / 48, w = l % 48;
    int lw = w * 48 + h;
    long base = (long)b * 4 * 128 * HWSZ;
    float v = y[base + (long)(0 * 128 + d) * HWSZ + l]
            + y[base + (long)(2 * 128 + d) * HWSZ + (2303 - l)]
            + y[base + (long)(1 * 128 + d) * HWSZ + lw]
            + y[base + (long)(3 * 128 + d) * HWSZ + (2303 - lw)];
    float s1 = v;
    for (int o = 16; o > 0; o >>= 1) s1 += __shfl_xor(s1, o, 32);
    int wid = d >> 5;
    if ((d & 31) == 0) sm[wid] = s1;
    __syncthreads();
    float mu = (sm[0] + sm[1] + sm[2] + sm[3]) / 128.0f;
    float dv = v - mu;
    float s2 = dv * dv;
    for (int o = 16; o > 0; o >>= 1) s2 += __shfl_xor(s2, o, 32);
    if ((d & 31) == 0) sq[wid] = s2;
    __syncthreads();
    float var = (sq[0] + sq[1] + sq[2] + sq[3]) / 128.0f;
    float yn = dv * rsqrtf(var + EPSV) * lng[d] + lnb[d];
    float zv = xz[(long)(128 + d) * NTOT + (long)b * HWSZ + l];
    gh[((long)b * HWSZ + l) * 128 + d] = (_Float16)(yn * siluf(zv));
}

// final: out(B,C,H,W) = fc2buf + fc2_b + x1
__global__ void final_k(const float* __restrict__ fc2buf, const float* __restrict__ fc2b,
                        const float* __restrict__ aout, float* __restrict__ out)
{
    int i = blockIdx.x * blockDim.x + threadIdx.x;
    if (i >= 64 * NTOT) return;
    int c = i / NTOT, bp = i % NTOT, b = bp / HWSZ, p = bp % HWSZ;
    out[((long)b * 64 + c) * HWSZ + p] = fc2buf[i] + fc2b[c] + aout[i];
}

// ---------------------------------------------------------------------------
extern "C" void kernel_launch(void* const* d_in, const int* in_sizes, int n_in,
                              void* d_out, int out_size, void* d_ws, size_t ws_size,
                              hipStream_t stream)
{
    (void)in_sizes; (void)n_in; (void)out_size; (void)ws_size;
    const float* x      = (const float*)d_in[0];
    const float* sg[5]  = {(const float*)d_in[1], (const float*)d_in[4], (const float*)d_in[7], (const float*)d_in[10], (const float*)d_in[13]};
    const float* sb[5]  = {(const float*)d_in[2], (const float*)d_in[5], (const float*)d_in[8], (const float*)d_in[11], (const float*)d_in[14]};
    const float* sw[5]  = {(const float*)d_in[3], (const float*)d_in[6], (const float*)d_in[9], (const float*)d_in[12], (const float*)d_in[15]};
    const float* comp_g = (const float*)d_in[16];
    const float* comp_b = (const float*)d_in[17];
    const float* comp_w = (const float*)d_in[18];
    const float* sc_g   = (const float*)d_in[19];
    const float* sc_b   = (const float*)d_in[20];
    const float* sc_w   = (const float*)d_in[21];
    const float* inpw   = (const float*)d_in[22];
    const float* dww    = (const float*)d_in[23];
    const float* dwb    = (const float*)d_in[24];
    const float* xprojw = (const float*)d_in[25];
    const float* dtw    = (const float*)d_in[26];
    const float* dtb    = (const float*)d_in[27];
    const float* alogs  = (const float*)d_in[28];
    const float* Ds     = (const float*)d_in[29];
    const float* lng    = (const float*)d_in[30];
    const float* lnb    = (const float*)d_in[31];
    const float* outw   = (const float*)d_in[32];
    const float* fc1w   = (const float*)d_in[33];
    const float* fc1b   = (const float*)d_in[34];
    const float* fc2w   = (const float*)d_in[35];
    const float* fc2b   = (const float*)d_in[36];
    float* outp = (float*)d_out;

    char* wsp = (char*)d_ws;
    size_t off = 0;
    auto alloc = [&](size_t bytes) -> void* {
        void* p = wsp + off; off = (off + bytes + 255) & ~(size_t)255; return p;
    };
    _Float16* h0   = (_Float16*)alloc((size_t)NTOT * 64 * 2);
    _Float16* h1   = (_Float16*)alloc((size_t)NTOT * 64 * 2);
    _Float16* h2   = (_Float16*)alloc((size_t)1152 * 64 * 2);
    _Float16* h3   = (_Float16*)alloc((size_t)288 * 64 * 2);
    _Float16* hsc  = (_Float16*)alloc((size_t)NTOT * 64 * 2);
    float* z0   = (float*)alloc((size_t)128 * NTOT * 4);
    float* z1   = (float*)alloc((size_t)128 * NTOT * 4);
    float* z2   = (float*)alloc((size_t)128 * 1152 * 4);
    float* z3   = (float*)alloc((size_t)128 * 288 * 4);
    float* z4   = (float*)alloc(256 * 4);
    _Float16* cath = (_Float16*)alloc((size_t)NTOT * 640 * 2);
    float* compo = (float*)alloc((size_t)64 * NTOT * 4);
    float* zsc   = (float*)alloc((size_t)64 * NTOT * 4);
    _Float16* ph = (_Float16*)alloc((size_t)NTOT * 64 * 2);
    float* xz    = (float*)alloc((size_t)256 * NTOT * 4);
    float* xcs   = (float*)alloc((size_t)128 * NTOT * 4);
    _Float16* xsh = (_Float16*)alloc((size_t)8 * HWSZ * 128 * 2);
    float* xdbl  = (float*)alloc((size_t)8 * 48 * HWSZ * 4);
    float* delta = (float*)alloc((size_t)8 * 128 * HWSZ * 4);
    float* acoef = (float*)alloc(4 * 128 * 16 * 4);
    float* ybuf  = (float*)alloc((size_t)8 * 128 * HWSZ * 4);
    _Float16* gh = (_Float16*)alloc((size_t)NTOT * 128 * 2);
    float* aout  = (float*)alloc((size_t)64 * NTOT * 4);
    _Float16* x1h = (_Float16*)alloc((size_t)NTOT * 64 * 2);
    _Float16* h1h = (_Float16*)alloc((size_t)NTOT * 128 * 2);
    float* fc2buf = (float*)alloc((size_t)64 * NTOT * 4);
    _Float16* swh[5];
    for (int j = 0; j < 5; ++j) swh[j] = (_Float16*)alloc(128 * 64 * 2);
    _Float16* scwh  = (_Float16*)alloc(64 * 64 * 2);
    _Float16* inwh  = (_Float16*)alloc(256 * 64 * 2);
    _Float16* xpwh  = (_Float16*)alloc(4 * 36 * 128 * 2);
    _Float16* outwh = (_Float16*)alloc(64 * 128 * 2);
    _Float16* fc1wh = (_Float16*)alloc(128 * 64 * 2);
    _Float16* fc2wh = (_Float16*)alloc(64 * 128 * 2);
    _Float16* compwh = (_Float16*)alloc((size_t)9 * 64 * 640 * 2);

    auto nb = [](int n, int t) { return (n + t - 1) / t; };
    // weight conversions
    for (int j = 0; j < 5; ++j)
        cvt16<<<nb(8192, 256), 256, 0, stream>>>(sw[j], swh[j], 8192);
    cvt16<<<nb(4096, 256), 256, 0, stream>>>(sc_w, scwh, 4096);
    cvt16<<<nb(16384, 256), 256, 0, stream>>>(inpw, inwh, 16384);
    cvt16<<<nb(4 * 36 * 128, 256), 256, 0, stream>>>(xprojw, xpwh, 4 * 36 * 128);
    cvt16<<<nb(8192, 256), 256, 0, stream>>>(outw, outwh, 8192);
    cvt16<<<nb(8192, 256), 256, 0, stream>>>(fc1w, fc1wh, 8192);
    cvt16<<<nb(8192, 256), 256, 0, stream>>>(fc2w, fc2wh, 8192);
    prep_compw<<<nb(9 * 64 * 640, 256), 256, 0, stream>>>(comp_w, compwh);
    acoef_k<<<nb(8192, 256), 256, 0, stream>>>(alogs, acoef);

    // PAPPM branch activations (pixel-major f16)
    bnrelu_x<<<nb(64 * NTOT, 256), 256, 0, stream>>>(x, sg[0], sb[0], h0);
    bnrelu_x<<<nb(64 * NTOT, 256), 256, 0, stream>>>(x, sc_g, sc_b, hsc);
    pool_bnrelu<<<nb(64 * NTOT, 256), 256, 0, stream>>>(x, sg[1], sb[1], h1, 3, 1, 1, 48, 64 * NTOT);
    pool_bnrelu<<<nb(64 * 1152, 256), 256, 0, stream>>>(x, sg[2], sb[2], h2, 5, 2, 2, 24, 64 * 1152);
    pool_bnrelu<<<nb(64 * 288, 256), 256, 0, stream>>>(x, sg[3], sb[3], h3, 9, 4, 4, 12, 64 * 288);
    mean_s4<<<2, 128, 0, stream>>>(x, sg[4], sb[4], sw[4], z4);

    auto gemm4 = [&](const _Float16* A, const _Float16* B, float* C, _Float16* ChT, int ldch,
                     const float* bias, int Mreal, int Mtiles, int K, int N, int act,
                     long sA, long sB, long sC, int Amod, int nbat) {
        dim3 g(N / 64, Mtiles, nbat);
        wmma_gemm<4><<<g, 32, 0, stream>>>(A, B, C, ChT, ldch, bias, Mreal, K, N, act, sA, sB, sC, Amod);
    };
    auto gemm1 = [&](const _Float16* A, const _Float16* B, float* C, _Float16* ChT, int ldch,
                     const float* bias, int Mreal, int Mtiles, int K, int N, int act) {
        dim3 g(N / 16, Mtiles, 1);
        wmma_gemm<1><<<g, 32, 0, stream>>>(A, B, C, ChT, ldch, bias, Mreal, K, N, act, 0, 0, 0, 1);
    };
    // branch 1x1 convs (GEMMs)
    gemm4(swh[0], h0, z0, nullptr, 0, nullptr, 128, 8, 64, NTOT, 0, 0, 0, 0, 1, 1);
    gemm4(swh[1], h1, z1, nullptr, 0, nullptr, 128, 8, 64, NTOT, 0, 0, 0, 0, 1, 1);
    gemm4(swh[2], h2, z2, nullptr, 0, nullptr, 128, 8, 64, 1152, 0, 0, 0, 0, 1, 1);
    gemm1(swh[3], h3, z3, nullptr, 0, nullptr, 128, 8, 64, 288, 0);
    gemm4(scwh, hsc, zsc, nullptr, 0, nullptr, 64, 4, 64, NTOT, 0, 0, 0, 0, 1, 1);

    assemble_cat<<<nb(128 * NTOT, 256), 256, 0, stream>>>(z0, z1, z2, z3, z4, comp_g, comp_b, cath);
    { dim3 g(36, 4, 2); wmma_conv3x3<<<g, 32, 0, stream>>>(compwh, cath, compo); }
    make_ph<<<nb(64 * NTOT, 256), 256, 0, stream>>>(compo, zsc, ph);

    // SS2D
    gemm4(inwh, ph, xz, nullptr, 0, nullptr, 256, 16, 64, NTOT, 0, 0, 0, 0, 1, 1);
    dwconv_silu<<<nb(128 * NTOT, 256), 256, 0, stream>>>(xz, dww, dwb, xcs);
    xs_fill<<<nb(8 * HWSZ * 128, 256), 256, 0, stream>>>(xcs, xsh);
    gemm4(xpwh, xsh, xdbl, nullptr, 0, nullptr, 36, 3, 128, HWSZ, 0,
          (long)36 * 128, (long)HWSZ * 128, (long)48 * HWSZ, 4, 8);
    delta_k<<<nb(8 * 128 * HWSZ, 256), 256, 0, stream>>>(xdbl, dtw, dtb, delta);
    { dim3 g(8, 4); scan_k<<<g, 512, 0, stream>>>(delta, xdbl, xcs, acoef, Ds, ybuf); }
    combine_ln<<<NTOT, 128, 0, stream>>>(ybuf, xz, lng, lnb, gh);
    gemm4(outwh, gh, aout, x1h, 64, nullptr, 64, 4, 128, NTOT, 0, 0, 0, 0, 1, 1);

    // MLP + residual
    gemm4(fc1wh, x1h, nullptr, h1h, 128, fc1b, 128, 8, 64, NTOT, 1, 0, 0, 0, 1, 1);
    gemm4(fc2wh, h1h, fc2buf, nullptr, 0, nullptr, 64, 4, 128, NTOT, 0, 0, 0, 0, 1, 1);
    final_k<<<nb(64 * NTOT, 256), 256, 0, stream>>>(fc2buf, fc2b, aout, outp);
}
